// Model_6863357739715
// MI455X (gfx1250) — matrix operand order repaired and checked by reading, not compiled, not run on hardware
//
#include <hip/hip_runtime.h>
#include <hip/hip_bf16.h>

typedef _Float16 v16h __attribute__((ext_vector_type(16)));
typedef _Float16 v8h  __attribute__((ext_vector_type(8)));
typedef _Float16 v4h  __attribute__((ext_vector_type(4)));
typedef _Float16 v2h  __attribute__((ext_vector_type(2)));
typedef float    v8f  __attribute__((ext_vector_type(8)));
typedef float    v4f  __attribute__((ext_vector_type(4)));

#define S_LEN 512
#define F_IN  64
#define HID   32
#define G4    128   // 4*H gate width
#define MT    16    // batch tile rows (WMMA M)
#define XST   72    // xh row stride (halfs), padded for bank spread (144B)
#define HST   40    // hh row stride (halfs), padded (80B)
#define ZST   132   // z row stride (floats), padded

__device__ __forceinline__ float fsig(float x) {
    // sigmoid(x) = 1 / (1 + 2^(-x*log2e))
    return __builtin_amdgcn_rcpf(1.0f + __builtin_amdgcn_exp2f(-1.44269504f * x));
}
__device__ __forceinline__ float ftanh(float x) {
    // tanh(x) = 1 - 2 / (2^(2x*log2e) + 1); saturates correctly at +-inf
    float e = __builtin_amdgcn_exp2f(2.88539008f * x);
    return 1.0f - 2.0f * __builtin_amdgcn_rcpf(e + 1.0f);
}

__global__ __launch_bounds__(256, 2)
void lstm_mlp_fused(const float* __restrict__ x,  const float* __restrict__ Wx,
                    const float* __restrict__ Wh, const float* __restrict__ b,
                    const float* __restrict__ W1, const float* __restrict__ b1,
                    const float* __restrict__ W2, const float* __restrict__ b2,
                    const float* __restrict__ Wo, const float* __restrict__ bo,
                    float* __restrict__ out)
{
    __shared__ __align__(16) _Float16 xh[2][MT * XST];  // x tile, f16, double-buffered
    __shared__ __align__(16) _Float16 hh[MT * HST];     // hidden state, f16
    __shared__ __align__(16) float    zsh[MT * ZST];    // pre-activation gates, f32

    const int tid  = threadIdx.x;
    const int lane = tid & 31;
    const int wv   = tid >> 5;              // 0..7 : which 16-col N-tile of the 128 gates
    const int b0   = blockIdx.x * MT;

    // ---- loop-invariant B fragments (Wx split into two K=32 halves, Wh one) ----
    // B layout (16x16x32 f16): lane holds column N = lane&15; lanes 0-15 cover
    // K=0..7,16..23, lanes 16-31 cover K=8..15,24..31 (matches A fragment order).
    const int ncol = wv * 16 + (lane & 15);
    const int krow = (lane >> 4) * 8;       // 0 or 8
    v16h bx0, bx1, bh;
    #pragma unroll
    for (int v = 0; v < 8; ++v) {
        int k = krow + 2 * v + (v >= 4 ? 8 : 0);
        bx0[2*v]   = (_Float16)Wx[(size_t)(k     ) * G4 + ncol];
        bx0[2*v+1] = (_Float16)Wx[(size_t)(k +  1) * G4 + ncol];
        bx1[2*v]   = (_Float16)Wx[(size_t)(k + 32) * G4 + ncol];
        bx1[2*v+1] = (_Float16)Wx[(size_t)(k + 33) * G4 + ncol];
        bh [2*v]   = (_Float16)Wh[(size_t)(k     ) * G4 + ncol];
        bh [2*v+1] = (_Float16)Wh[(size_t)(k +  1) * G4 + ncol];
    }
    // Column bias folded into one WMMA's C operand: the 16x16 f32 C/D layout holds
    // column `ncol` in every one of a lane's 8 VGPRs, so a splat is a legal C init.
    const float bias = b[ncol];
    v8f cbias;
    #pragma unroll
    for (int r = 0; r < 8; ++r) cbias[r] = bias;

    // ---- role mappings ----
    const int am  = lane & 15;              // A-fragment row M
    const int akb = (lane >> 4) * 8;        // A-fragment K base (16-bit A 16x32 layout)
    const int xrow = tid >> 4;              // x staging: row 0..15
    const int xf   = (tid & 15) * 4;        //            4 floats per thread
    const int grow = tid >> 4;              // gate math: row 0..15
    const int gj   = tid & 15;              //            element pair 2*gj, 2*gj+1

    // ---- init: h0 = 0, stage x[t=0] ----
    for (int i = tid; i < MT * HST; i += 256) hh[i] = (_Float16)0.0f;
    {
        v4f xv = *(const v4f*)(x + ((size_t)(b0 + xrow) * S_LEN + 0) * F_IN + xf);
        v4h xc = { (_Float16)xv.x, (_Float16)xv.y, (_Float16)xv.z, (_Float16)xv.w };
        *(v4h*)&xh[0][xrow * XST + xf] = xc;
    }
    __syncthreads();

    float c0 = 0.0f, c1 = 0.0f;             // cell state, 2 elements per thread

    for (int t = 0; t < S_LEN; ++t) {
        // -- prefetch next timestep's x tile (latency hidden under this step) --
        const int tn = (t + 1 < S_LEN) ? (t + 1) : t;
        v4f xnext = *(const v4f*)(x + ((size_t)(b0 + xrow) * S_LEN + tn) * F_IN + xf);

        // -- A fragments from LDS (contiguous 16B chunks per the 16-bit A layout) --
        const _Float16* xp = &xh[t & 1][am * XST];
        const _Float16* hp = &hh[am * HST];
        union { v16h v; v8h h[2]; } ax0, ax1, ah;
        ax0.h[0] = *(const v8h*)(xp + akb);
        ax0.h[1] = *(const v8h*)(xp + akb + 16);
        ax1.h[0] = *(const v8h*)(xp + akb + 32);
        ax1.h[1] = *(const v8h*)(xp + akb + 48);
        ah.h[0]  = *(const v8h*)(hp + akb);
        ah.h[1]  = *(const v8h*)(hp + akb + 16);

        // -- z = x_t@Wx + h@Wh + b : three INDEPENDENT accumulators so the three
        //    WMMAs have no D->C register dependence (no hazard NOPs between them);
        //    bias rides in as the C operand of the h-GEMM.
        v8f acc0 = {};
        v8f acc1 = {};
        acc0 = __builtin_amdgcn_wmma_f32_16x16x32_f16(false, ax0.v, false, bx0, (short)0, acc0,  false, false);
        acc1 = __builtin_amdgcn_wmma_f32_16x16x32_f16(false, ax1.v, false, bx1, (short)0, acc1,  false, false);
        v8f acch = __builtin_amdgcn_wmma_f32_16x16x32_f16(false, ah.v, false, bh, (short)0, cbias, false, false);

        // -- combine partials + scatter z to LDS per the 16x16 f32 C/D layout --
        {
            const int rbase = (lane >> 4) * 8;
            #pragma unroll
            for (int r = 0; r < 8; ++r)
                zsh[(rbase + r) * ZST + ncol] = (acc0[r] + acc1[r]) + acch[r];
        }
        __syncthreads();

        // -- gates + state update: this thread owns elements (grow, 2gj) & (grow, 2gj+1) --
        {
            const float* zr = &zsh[grow * ZST + 2 * gj];
            float i0 = fsig(zr[0]),   i1 = fsig(zr[1]);
            float f0 = fsig(zr[32]),  f1 = fsig(zr[33]);
            float g0 = ftanh(zr[64]), g1 = ftanh(zr[65]);
            float o0 = fsig(zr[96]),  o1 = fsig(zr[97]);
            c0 = f0 * c0 + i0 * g0;
            c1 = f1 * c1 + i1 * g1;
            float h0 = o0 * ftanh(c0);
            float h1 = o1 * ftanh(c1);
            v2h hc = { (_Float16)h0, (_Float16)h1 };
            *(v2h*)&hh[grow * HST + 2 * gj] = hc;
        }

        // -- commit prefetched x into the other buffer --
        {
            v4h xc = { (_Float16)xnext.x, (_Float16)xnext.y, (_Float16)xnext.z, (_Float16)xnext.w };
            *(v4h*)&xh[(t + 1) & 1][xrow * XST + xf] = xc;
        }
        __syncthreads();
    }

    // ---- MLP head: 32 -> 32 -> 16 -> 3, leaky_relu(0.01). Tiny: 16 threads/WG. ----
    if (tid < MT) {
        float hv[HID];
        #pragma unroll
        for (int k = 0; k < HID; ++k) hv[k] = (float)hh[tid * HST + k];
        float a1[32];
        for (int j = 0; j < 32; ++j) {
            float s = b1[j];
            #pragma unroll
            for (int k = 0; k < 32; ++k) s += hv[k] * W1[k * 32 + j];
            a1[j] = (s > 0.0f) ? s : 0.01f * s;
        }
        float a2[16];
        for (int j = 0; j < 16; ++j) {
            float s = b2[j];
            #pragma unroll
            for (int k = 0; k < 32; ++k) s += a1[k] * W2[k * 16 + j];
            a2[j] = (s > 0.0f) ? s : 0.01f * s;
        }
        #pragma unroll
        for (int o = 0; o < 3; ++o) {
            float s = bo[o];
            #pragma unroll
            for (int k = 0; k < 16; ++k) s += a2[k] * Wo[k * 3 + o];
            out[(size_t)(b0 + tid) * 3 + o] = s;
        }
    }
}

extern "C" void kernel_launch(void* const* d_in, const int* in_sizes, int n_in,
                              void* d_out, int out_size, void* d_ws, size_t ws_size,
                              hipStream_t stream) {
    (void)n_in; (void)out_size; (void)d_ws; (void)ws_size;
    const float* x  = (const float*)d_in[0];
    const float* Wx = (const float*)d_in[1];
    const float* Wh = (const float*)d_in[2];
    const float* b  = (const float*)d_in[3];
    const float* W1 = (const float*)d_in[4];
    const float* b1 = (const float*)d_in[5];
    const float* W2 = (const float*)d_in[6];
    const float* b2 = (const float*)d_in[7];
    const float* Wo = (const float*)d_in[8];
    const float* bo = (const float*)d_in[9];
    float* out = (float*)d_out;

    const int B = in_sizes[0] / (S_LEN * F_IN);   // 2048
    const int nblocks = B / MT;                   // 128 workgroups x 8 waves
    lstm_mlp_fused<<<dim3(nblocks), dim3(256), 0, stream>>>(
        x, Wx, Wh, b, W1, b1, W2, b2, Wo, bo, out);
}
